// FourierSelfAttention_1984274891403
// MI455X (gfx1250) — matrix operand order repaired and checked by reading, not compiled, not run on hardware
//
#include <hip/hip_runtime.h>
#include <hip/hip_bf16.h>

// ---------------------------------------------------------------------------
// Fourier self-attention for MI455X (gfx1250, wave32, WMMA + TDM).
// Pipeline:
//   1) cast x / w_qkv / w_out fp32 -> bf16
//   2) QKV GEMM  (bf16 WMMA, f32 acc, TDM double-buffered LDS staging)
//      -> qkvf planes [3][B][H][Dh][T] f32
//   3) forward rfft (fp32 Stockham in LDS), packed in place over qkvf rows
//   4) freq-attention: logits over Dh, softmax over F=2049, scale v_fft;
//      result overwrites the (dead) q-plane
//   5) inverse rfft -> bf16 activation [B,T,C]
//   6) output GEMM (bf16 WMMA + TDM) + bias -> d_out (f32)
// ---------------------------------------------------------------------------

typedef __bf16 bf16_t;
typedef __attribute__((ext_vector_type(16))) __bf16 v16bf;
typedef __attribute__((ext_vector_type(8)))  __bf16 v8bf;
typedef __attribute__((ext_vector_type(8)))  float  v8f;
typedef __attribute__((ext_vector_type(4)))  unsigned int v4u;
typedef __attribute__((ext_vector_type(8)))  int    v8i;
typedef __attribute__((ext_vector_type(4)))  int    v4i;

#define B_   4
#define T_   4096
#define C_   1024
#define H_   32
#define DH_  32
#define M_   (B_ * T_)       // 16384 rows
#define NQKV (3 * C_)        // 3072
#define PLANE ((size_t)B_ * H_ * DH_ * T_)   // floats per q/k/v plane

// ---------------- helpers ----------------

__device__ inline bf16_t f2bf(float x) {
  union { float f; unsigned u; } v; v.f = x;
  unsigned r = v.u + 0x7FFFu + ((v.u >> 16) & 1u);   // round-to-nearest-even
  unsigned short hs = (unsigned short)(r >> 16);
  union { unsigned short s; bf16_t b; } o; o.s = hs;
  return o.b;
}

__device__ inline float2 cadd(float2 a, float2 b) { return make_float2(a.x + b.x, a.y + b.y); }
__device__ inline float2 csub(float2 a, float2 b) { return make_float2(a.x - b.x, a.y - b.y); }

__global__ void cast_f32_bf16(const float* __restrict__ in, bf16_t* __restrict__ out, int n) {
  int i = blockIdx.x * blockDim.x + threadIdx.x;
  int stride = gridDim.x * blockDim.x;
  for (; i < n; i += stride) out[i] = f2bf(in[i]);
}

// ---------------- Tensor Data Mover: 2D tile global -> LDS ----------------
// Builds D# group0/group1 per CDNA5 ISA 08_async_tensor §8 and issues
// TENSOR_LOAD_TO_LDS (tracked by TENSORcnt). 2D tensor: groups 2/3 unused.
//   group0: [1:0]=count=1 | [63:32]=lds_addr | [120:64]=global_addr | type=2
//   group1: data_size=1(2B), tensor_dim0=tile_x, tensor_dim1=tile_y,
//           tile_dim0=tile_x, tile_dim1=tile_y, tensor_dim0_stride=row stride
__device__ inline void tdm_load_2d_bf16(const void* gptr, unsigned lds_off,
                                        unsigned tile_x, unsigned tile_y,
                                        unsigned stride_elems) {
  unsigned long long ga = (unsigned long long)(size_t)gptr;
  v4u g0;
  g0[0] = 1u;                                        // count=1, user descriptor
  g0[1] = lds_off;                                   // LDS byte address
  g0[2] = (unsigned)ga;                              // global_addr[31:0]
  g0[3] = (unsigned)((ga >> 32) & 0x01FFFFFFull)     // global_addr[56:32]
          | (2u << 30);                              // type=2 ("image")
  v8i g1;
  g1[0] = (int)(1u << 16);                           // data_size=1 -> 2 bytes
  g1[1] = (int)((tile_x & 0xFFFFu) << 16);           // tensor_dim0[15:0]
  g1[2] = (int)(((tile_x >> 16) & 0xFFFFu)           // tensor_dim0[31:16]
          | ((tile_y & 0xFFFFu) << 16));             // tensor_dim1[15:0]
  g1[3] = (int)(((tile_y >> 16) & 0xFFFFu)           // tensor_dim1[31:16]
          | ((tile_x & 0xFFFFu) << 16));             // tile_dim0
  g1[4] = (int)(tile_y & 0xFFFFu);                   // tile_dim1 (tile_dim2=0)
  g1[5] = (int)stride_elems;                         // tensor_dim0_stride[31:0]
  g1[6] = 0;                                         // stride[47:32], dim1_stride lo
  g1[7] = 0;
  v4i zz = {0, 0, 0, 0};
#if defined(__clang_major__) && (__clang_major__ >= 23)
  v8i z8 = {0, 0, 0, 0, 0, 0, 0, 0};
  __builtin_amdgcn_tensor_load_to_lds(g0, g1, zz, zz, z8, 0);
#else
  __builtin_amdgcn_tensor_load_to_lds(g0, g1, zz, zz, 0);
#endif
}

// ---------------- WMMA fragment loads (per CDNA5 ISA 7.12.2) ----------------
// A (16-bit, 16x32): lane L: m = L%16, half = L/16.
//   frag[e]   = row[half*8 + e]; frag[8+e] = row[16 + half*8 + e]
__device__ inline v16bf load_fragA(const bf16_t* rowp, int h) {
  v8bf lo = *(const v8bf*)(rowp + h * 8);
  v8bf hi = *(const v8bf*)(rowp + 16 + h * 8);
  v16bf f;
#pragma unroll
  for (int e = 0; e < 8; e++) { f[e] = lo[e]; f[e + 8] = hi[e]; }
  return f;
}
// B (32x16): lane L: n = L%16, half = L/16; frag[e] = K half*8+e, frag[8+e] =
// K 16+half*8+e of W row n (Y = X * W^T).
__device__ inline v16bf load_fragB(const bf16_t* rowp, int h) {
  v8bf lo = *(const v8bf*)(rowp + h * 8);
  v8bf hi = *(const v8bf*)(rowp + 16 + h * 8);
  v16bf f;
#pragma unroll
  for (int e = 0; e < 8; e++) { f[e] = lo[e]; f[e + 8] = hi[e]; }
  return f;
}

// ---------------- GEMM cores ----------------
// Block tile 128x128, 8 waves (4x2), wave tile 32x64 -> 8 WMMA / wave / K-step.
// A/B tiles staged by the TDM, double-buffered; wave 0 issues + drains
// TENSORcnt, workgroup barrier publishes the buffer.

#define BM 128
#define BN 128
#define BK 32

__global__ __launch_bounds__(256) void gemm_qkv_wmma(
    const bf16_t* __restrict__ A,   // [M_, C_] row-major bf16
    const bf16_t* __restrict__ W,   // [NQKV, C_] row-major bf16
    float* __restrict__ qkvf)       // [3][B][H][Dh][T] f32
{
  __shared__ __align__(16) bf16_t As[2][BM][BK];
  __shared__ __align__(16) bf16_t Bs[2][BN][BK];
  const int K = C_;
  const int tid  = threadIdx.x;
  const int bm   = blockIdx.x, bn = blockIdx.y;
  const int lane = tid & 31, wave = tid >> 5;
  const int wm   = wave >> 1, wn = wave & 1;       // 4x2 waves, 32x64 each
  const int r    = lane & 15, h = lane >> 4;
  const bf16_t* Abase = A + (size_t)(bm * BM) * K;
  const bf16_t* Bbase = W + (size_t)(bn * BN) * K;

  if (wave == 0) {   // prime buffer 0 (TDM ignores EXEC; scalar-branch per wave)
    tdm_load_2d_bf16(Abase, (unsigned)(size_t)&As[0][0][0], BK, BM, K);
    tdm_load_2d_bf16(Bbase, (unsigned)(size_t)&Bs[0][0][0], BK, BN, K);
    __builtin_amdgcn_s_wait_tensorcnt(0);
  }
  __syncthreads();

  v8f acc[2][4] = {};
  int buf = 0;
  for (int k0 = 0; k0 < K; k0 += BK) {
    int nbuf = buf ^ 1;
    if (wave == 0 && k0 + BK < K) {   // stream next K-tile while computing
      tdm_load_2d_bf16(Abase + (k0 + BK), (unsigned)(size_t)&As[nbuf][0][0], BK, BM, K);
      tdm_load_2d_bf16(Bbase + (k0 + BK), (unsigned)(size_t)&Bs[nbuf][0][0], BK, BN, K);
    }
    v16bf af[2], bfg[4];
#pragma unroll
    for (int i = 0; i < 2; i++) af[i]  = load_fragA(&As[buf][wm * 32 + i * 16 + r][0], h);
#pragma unroll
    for (int j = 0; j < 4; j++) bfg[j] = load_fragB(&Bs[buf][wn * 64 + j * 16 + r][0], h);
#pragma unroll
    for (int i = 0; i < 2; i++)
#pragma unroll
      for (int j = 0; j < 4; j++)
        acc[i][j] = __builtin_amdgcn_wmma_f32_16x16x32_bf16(
            false, af[i], false, bfg[j], (short)0, acc[i][j], false, false);
    if (wave == 0 && k0 + BK < K) __builtin_amdgcn_s_wait_tensorcnt(0);
    __syncthreads();
    buf = nbuf;
  }
  // Epilogue: lane n = r, rows m = h*8 + e (8 consecutive) -> contiguous run
  // along T in the transposed head plane [3][B][H][Dh][T].
#pragma unroll
  for (int i = 0; i < 2; i++)
#pragma unroll
    for (int j = 0; j < 4; j++) {
      int o = bn * BN + wn * 64 + j * 16 + r;
      int m = bm * BM + wm * 32 + i * 16 + h * 8;
      int which = o >> 10, rem = o & 1023;
      int hh = rem >> 5, dd = rem & 31;
      int b = m >> 12, t = m & 4095;
      float* p = qkvf + ((((size_t)which * B_ + b) * H_ + hh) * DH_ + dd) * (size_t)T_ + t;
      v8f c = acc[i][j];
      *(float4*)(p)     = make_float4(c[0], c[1], c[2], c[3]);
      *(float4*)(p + 4) = make_float4(c[4], c[5], c[6], c[7]);
    }
}

__global__ __launch_bounds__(256) void gemm_out_wmma(
    const bf16_t* __restrict__ A,   // [M_, C_] bf16 activations
    const bf16_t* __restrict__ W,   // [C_, C_] bf16
    const float* __restrict__ bias, // [C_]
    float* __restrict__ out)        // [M_, C_] f32
{
  __shared__ __align__(16) bf16_t As[2][BM][BK];
  __shared__ __align__(16) bf16_t Bs[2][BN][BK];
  const int K = C_;
  const int tid  = threadIdx.x;
  const int bm   = blockIdx.x, bn = blockIdx.y;
  const int lane = tid & 31, wave = tid >> 5;
  const int wm   = wave >> 1, wn = wave & 1;
  const int r    = lane & 15, h = lane >> 4;
  const bf16_t* Abase = A + (size_t)(bm * BM) * K;
  const bf16_t* Bbase = W + (size_t)(bn * BN) * K;

  if (wave == 0) {
    tdm_load_2d_bf16(Abase, (unsigned)(size_t)&As[0][0][0], BK, BM, K);
    tdm_load_2d_bf16(Bbase, (unsigned)(size_t)&Bs[0][0][0], BK, BN, K);
    __builtin_amdgcn_s_wait_tensorcnt(0);
  }
  __syncthreads();

  v8f acc[2][4] = {};
  int buf = 0;
  for (int k0 = 0; k0 < K; k0 += BK) {
    int nbuf = buf ^ 1;
    if (wave == 0 && k0 + BK < K) {
      tdm_load_2d_bf16(Abase + (k0 + BK), (unsigned)(size_t)&As[nbuf][0][0], BK, BM, K);
      tdm_load_2d_bf16(Bbase + (k0 + BK), (unsigned)(size_t)&Bs[nbuf][0][0], BK, BN, K);
    }
    v16bf af[2], bfg[4];
#pragma unroll
    for (int i = 0; i < 2; i++) af[i]  = load_fragA(&As[buf][wm * 32 + i * 16 + r][0], h);
#pragma unroll
    for (int j = 0; j < 4; j++) bfg[j] = load_fragB(&Bs[buf][wn * 64 + j * 16 + r][0], h);
#pragma unroll
    for (int i = 0; i < 2; i++)
#pragma unroll
      for (int j = 0; j < 4; j++)
        acc[i][j] = __builtin_amdgcn_wmma_f32_16x16x32_bf16(
            false, af[i], false, bfg[j], (short)0, acc[i][j], false, false);
    if (wave == 0 && k0 + BK < K) __builtin_amdgcn_s_wait_tensorcnt(0);
    __syncthreads();
    buf = nbuf;
  }
#pragma unroll
  for (int i = 0; i < 2; i++)
#pragma unroll
    for (int j = 0; j < 4; j++) {
      int o  = bn * BN + wn * 64 + j * 16 + r;
      int m0 = bm * BM + wm * 32 + i * 16 + h * 8;
      float bv = bias[o];
      v8f c = acc[i][j];
#pragma unroll
      for (int e = 0; e < 8; e++)
        out[(size_t)(m0 + e) * C_ + o] = c[e] + bv;
    }
}

// ---------------- Stockham radix-2 FFT, N=4096, fp32 in LDS ----------------
__device__ inline int stockham4096(float2 (*buf)[4096], int tid, float sign) {
  int src = 0;
  for (int Ns = 1; Ns < 4096; Ns <<= 1) {
    for (int jj = tid; jj < 2048; jj += 256) {
      float2 v0 = buf[src][jj];
      float2 v1 = buf[src][jj + 2048];
      int jm = jj & (Ns - 1);
      float ang = sign * 3.14159265358979323846f * (float)jm / (float)Ns;
      float s, c;
      __sincosf(ang, &s, &c);
      float2 w1 = make_float2(v1.x * c - v1.y * s, v1.x * s + v1.y * c);
      int idx = ((jj - jm) << 1) + jm;
      buf[1 - src][idx]      = cadd(v0, w1);
      buf[1 - src][idx + Ns] = csub(v0, w1);
    }
    __syncthreads();
    src ^= 1;
  }
  return src;
}

// Forward rfft of one real row (length 4096), packed in place:
// slot0 = (X[0].re, X[2048].re); slots 1..2047 = X[f].
__global__ __launch_bounds__(256) void fft_fwd_kernel(float* __restrict__ qkvf) {
  __shared__ float2 buf[2][4096];
  int row = blockIdx.x;                       // 0 .. 3*B*H*Dh-1
  float* data = qkvf + (size_t)row * 4096;
  int tid = threadIdx.x;
  for (int i = tid; i < 4096; i += 256) buf[0][i] = make_float2(data[i], 0.f);
  __syncthreads();
  int src = stockham4096(buf, tid, -1.f);
  float x0 = buf[src][0].x;
  float xn = buf[src][2048].x;
  float2* outp = (float2*)data;
  for (int f = tid; f < 2048; f += 256) {
    float2 v = buf[src][f];
    if (f == 0) v = make_float2(x0, xn);
    outp[f] = v;
  }
}

// ---------------- frequency-domain attention + softmax ----------------
__global__ __launch_bounds__(256) void attn_kernel(
    float* __restrict__ qkvf, const float* __restrict__ scale_ptr) {
  __shared__ float attn[2049];
  __shared__ float red[256];
  int bh = blockIdx.x;                         // b*H + h
  float*       q = qkvf + (size_t)bh * DH_ * T_;            // also output plane
  const float* k = qkvf + PLANE     + (size_t)bh * DH_ * T_;
  const float* v = qkvf + 2 * PLANE + (size_t)bh * DH_ * T_;
  int tid = threadIdx.x;
  float scale = *scale_ptr;

  for (int f = tid; f < 2049; f += 256) {
    float s = 0.f;
    if (f == 0) {
      for (int d = 0; d < DH_; d++) s += q[(size_t)d * T_] * k[(size_t)d * T_];
    } else if (f == 2048) {
      for (int d = 0; d < DH_; d++) s += q[(size_t)d * T_ + 1] * k[(size_t)d * T_ + 1];
    } else {
      for (int d = 0; d < DH_; d++) {
        float2 qv = *(const float2*)(q + (size_t)d * T_ + 2 * f);
        float2 kv = *(const float2*)(k + (size_t)d * T_ + 2 * f);
        s += qv.x * kv.x + qv.y * kv.y;
      }
    }
    attn[f] = s * scale;
  }
  __syncthreads();
  float lm = -1e30f;
  for (int f = tid; f < 2049; f += 256) lm = fmaxf(lm, attn[f]);
  red[tid] = lm; __syncthreads();
  for (int s = 128; s > 0; s >>= 1) { if (tid < s) red[tid] = fmaxf(red[tid], red[tid + s]); __syncthreads(); }
  float gmax = red[0]; __syncthreads();
  float ls = 0.f;
  for (int f = tid; f < 2049; f += 256) { float e = __expf(attn[f] - gmax); attn[f] = e; ls += e; }
  red[tid] = ls; __syncthreads();
  for (int s = 128; s > 0; s >>= 1) { if (tid < s) red[tid] += red[tid + s]; __syncthreads(); }
  float inv = 1.0f / red[0]; __syncthreads();
  for (int f = tid; f < 2049; f += 256) attn[f] *= inv;
  __syncthreads();
  for (int idx = tid; idx < DH_ * 2048; idx += 256) {
    int d = idx >> 11, f = idx & 2047;
    float2 vv = *(const float2*)(v + (size_t)d * T_ + 2 * f);
    float2 ov = (f == 0) ? make_float2(attn[0] * vv.x, attn[2048] * vv.y)
                         : make_float2(attn[f] * vv.x, attn[f] * vv.y);
    *(float2*)(q + (size_t)d * T_ + 2 * f) = ov;
  }
}

// ---------------- inverse rfft -> bf16 activation [B,T,C] ----------------
__global__ __launch_bounds__(256) void fft_inv_kernel(
    const float* __restrict__ ofp, bf16_t* __restrict__ ybf) {
  __shared__ float2 buf[2][4096];
  int col = blockIdx.x;                        // ((b*H+h)*Dh + d)
  const float2* spec = (const float2*)(ofp + (size_t)col * 4096);
  int tid = threadIdx.x;
  for (int f = tid; f < 2048; f += 256) {
    float2 s = spec[f];
    if (f == 0) {
      buf[0][0]    = make_float2(s.x, 0.f);
      buf[0][2048] = make_float2(s.y, 0.f);
    } else {
      buf[0][f]        = s;
      buf[0][4096 - f] = make_float2(s.x, -s.y);   // Hermitian extension
    }
  }
  __syncthreads();
  int src = stockham4096(buf, tid, +1.f);
  int b = col >> 10;
  int c = col & 1023;                          // h*32 + d
  const float invN = 1.0f / 4096.0f;
  for (int t = tid; t < 4096; t += 256) {
    float y = buf[src][t].x * invN;
    ybf[(size_t)(b * 4096 + t) * C_ + c] = f2bf(y);
  }
}

// ---------------- launcher ----------------
extern "C" void kernel_launch(void* const* d_in, const int* in_sizes, int n_in,
                              void* d_out, int out_size, void* d_ws, size_t ws_size,
                              hipStream_t stream) {
  (void)in_sizes; (void)n_in; (void)out_size; (void)ws_size;
  const float* x      = (const float*)d_in[0];
  const float* w_qkv  = (const float*)d_in[1];
  const float* scale  = (const float*)d_in[2];
  const float* w_out  = (const float*)d_in[3];
  const float* b_out  = (const float*)d_in[4];
  float* out = (float*)d_out;

  char* ws = (char*)d_ws;
  size_t off = 0;
  bf16_t* xbf    = (bf16_t*)(ws + off); off += (size_t)M_ * C_ * 2;      // 32 MB
  bf16_t* wqkvbf = (bf16_t*)(ws + off); off += (size_t)NQKV * C_ * 2;    //  6 MB
  bf16_t* woutbf = (bf16_t*)(ws + off); off += (size_t)C_ * C_ * 2;      //  2 MB
  float*  qkvf   = (float*)(ws + off);  off += 3 * PLANE * 4;            // 192 MB
  bf16_t* ybf    = (bf16_t*)(ws + off); off += (size_t)M_ * C_ * 2;      // 32 MB

  cast_f32_bf16<<<2048, 256, 0, stream>>>(x, xbf, M_ * C_);
  cast_f32_bf16<<<1024, 256, 0, stream>>>(w_qkv, wqkvbf, NQKV * C_);
  cast_f32_bf16<<<512, 256, 0, stream>>>(w_out, woutbf, C_ * C_);

  gemm_qkv_wmma<<<dim3(M_ / BM, NQKV / BN), 256, 0, stream>>>(xbf, wqkvbf, qkvf);

  fft_fwd_kernel<<<3 * B_ * H_ * DH_, 256, 0, stream>>>(qkvf);

  attn_kernel<<<B_ * H_, 256, 0, stream>>>(qkvf, scale);

  fft_inv_kernel<<<B_ * H_ * DH_, 256, 0, stream>>>(qkvf, ybf);

  gemm_out_wmma<<<dim3(M_ / BM, C_ / BN), 256, 0, stream>>>(ybf, woutbf, b_out, out);
}
